// HGAT_22136261444132
// MI455X (gfx1250) — matrix operand order repaired and checked by reading, not compiled, not run on hardware
//
#include <hip/hip_runtime.h>

// ---------------------------------------------------------------------------
// HGAT (hyperbolic GAT) forward, 2 layers, N=8192, D=128, on gfx1250.
// Heavy path: fused masked softmax + aggregation using v_wmma_f32_16x16x32_f16.
// ---------------------------------------------------------------------------

typedef __attribute__((ext_vector_type(16))) _Float16 v16h;
typedef __attribute__((ext_vector_type(8)))  float    v8f;
typedef __attribute__((ext_vector_type(4)))  _Float16 h4;

#define NN 8192
#define DD 128
#define SPLIT 4
#define CHUNK (NN / SPLIT)          // 2048 columns per split
#define MINF   1e-15f
#define MAXNRM 0.996f               // 1 - EPS, EPS = 4e-3
#define ATCLIP 0.99999f             // 1 - 1e-5
#define NEGINF (-1e9f)

__device__ __forceinline__ float wsum(float v) {
#pragma unroll
  for (int o = 16; o > 0; o >>= 1) v += __shfl_xor(v, o, 32);
  return v;
}

// A-matrix f16 16x32 layout: lane<16 -> K = {0..7, 16..23}; lane>=16 -> {8..15, 24..31}
__device__ __forceinline__ v16h ldA16(const _Float16* p) {
  union { uint4 u[2]; v16h v; } t;
  t.u[0] = *(const uint4*)p;        // 8 halfs
  t.u[1] = *(const uint4*)(p + 16); // 8 halfs, +16 elements
  return t.v;
}
// Contiguous-K loader (unused): lane n holds 16 contiguous K (lo lanes K0..15, hi lanes K16..31)
__device__ __forceinline__ v16h ldB16(const _Float16* p) {
  union { uint4 u[2]; v16h v; } t;
  t.u[0] = *(const uint4*)p;
  t.u[1] = *(const uint4*)(p + 8);
  return t.v;
}

// ---------------- encode: h = proj(expmap0(x)), + f16 copy -------------------
__global__ void hgat_encode(const float* __restrict__ x, float* __restrict__ h,
                            _Float16* __restrict__ h16) {
  int lane = threadIdx.x & 31, wid = threadIdx.x >> 5;
  int row = blockIdx.x * 4 + wid;
  const float* xr = x + (size_t)row * DD + lane * 4;
  float v[4] = {xr[0], xr[1], xr[2], xr[3]};
  float n = fmaxf(sqrtf(wsum(v[0]*v[0] + v[1]*v[1] + v[2]*v[2] + v[3]*v[3])), MINF);
  float f = fminf(tanhf(n), MAXNRM) / n;
  float* hr = h + (size_t)row * DD + lane * 4;
  h4 o;
#pragma unroll
  for (int i = 0; i < 4; i++) { float r = f * v[i]; hr[i] = r; o[i] = (_Float16)r; }
  *(h4*)(h16 + (size_t)row * DD + lane * 4) = o;
}

// ------------- per-layer prep: W -> f16, bh = proj(expmap0(b)) --------------
__global__ void hgat_prep(const float* __restrict__ W, const float* __restrict__ b,
                          _Float16* __restrict__ wf16, float* __restrict__ bh) {
  __shared__ float sb[DD];
  __shared__ float snorm;
  int tid = threadIdx.x;
  float bv = b[tid];
  sb[tid] = bv * bv;
  __syncthreads();
  if (tid == 0) {
    float s = 0.f;
    for (int i = 0; i < DD; i++) s += sb[i];
    snorm = fmaxf(sqrtf(s), MINF);
  }
  __syncthreads();
  float n = snorm;
  float t = fminf(tanhf(n), MAXNRM);
  bh[tid] = t / n * bv;
  if (tid == 0) bh[DD] = t * t;      // squared norm of bh
  for (int k = tid; k < DD * DD; k += DD) wf16[k] = (_Float16)W[k];
}

// ---------------- mx = h @ W^T  (f16 WMMA, f32 accumulate) ------------------
__global__ void __launch_bounds__(256) hgat_gemm(const _Float16* __restrict__ A,
                                                 const _Float16* __restrict__ Bw,
                                                 float* __restrict__ C) {
  int lane = threadIdx.x & 31, wid = threadIdx.x >> 5;
  int rb = blockIdx.x * 16;
  int cb = wid * 16;                // 8 waves cover all 128 output dims
  int m = lane & 15;
  int koffA = (lane & 16) ? 8 : 0;
  int koffB = (lane & 16) ? 8 : 0;
  v8f c = {0.f,0.f,0.f,0.f,0.f,0.f,0.f,0.f};
#pragma unroll
  for (int kb = 0; kb < DD; kb += 32) {
    v16h a  = ldA16(A  + (size_t)(rb + m) * DD + kb + koffA);
    v16h bm = ldA16(Bw + (size_t)(cb + m) * DD + kb + koffB);
    c = __builtin_amdgcn_wmma_f32_16x16x32_f16(false, a, false, bm, (short)0, c, false, false);
  }
  int mh = (lane & 16) ? 8 : 0;
#pragma unroll
  for (int j = 0; j < 8; j++)
    C[(size_t)(rb + j + mh) * DD + cb + m] = c[j];
}

// ---- row ops: mobius matvec scale + mobius_add bias + proj + logmap0 -------
// writes xt transposed (f16, [DD][NN]) for WMMA B loads, plus el/er scalars
__global__ void hgat_rowops(const float* __restrict__ h, const float* __restrict__ mx,
                            const float* __restrict__ bh, const float* __restrict__ al,
                            const float* __restrict__ ar, _Float16* __restrict__ xtT,
                            float* __restrict__ el, float* __restrict__ er) {
  int lane = threadIdx.x & 31, wid = threadIdx.x >> 5;
  int row = blockIdx.x * 4 + wid;
  int d0 = lane * 4;
  const float* hr = h  + (size_t)row * DD + d0;
  const float* mr = mx + (size_t)row * DD + d0;
  float hv[4], mv[4];
#pragma unroll
  for (int i = 0; i < 4; i++) { hv[i] = hr[i]; mv[i] = mr[i]; }
  float xn  = fmaxf(sqrtf(wsum(hv[0]*hv[0]+hv[1]*hv[1]+hv[2]*hv[2]+hv[3]*hv[3])), MINF);
  float mxn = fmaxf(sqrtf(wsum(mv[0]*mv[0]+mv[1]*mv[1]+mv[2]*mv[2]+mv[3]*mv[3])), MINF);
  float xc  = fminf(xn, ATCLIP);
  float s   = tanhf(mxn / xn * atanhf(xc));
  float sn  = fminf(s, MAXNRM);                  // norm after proj
  float g   = sn / mxn;
  float x2  = sn * sn;
  float y2  = bh[DD];
  float hm[4], bv[4];
#pragma unroll
  for (int i = 0; i < 4; i++) { hm[i] = g * mv[i]; bv[i] = bh[d0 + i]; }
  float xy = wsum(hm[0]*bv[0] + hm[1]*bv[1] + hm[2]*bv[2] + hm[3]*bv[3]);
  float cx  = 1.f + 2.f * xy + y2;
  float cy  = 1.f - x2;
  float den = fmaxf(1.f + 2.f * xy + x2 * y2, MINF);
  float res[4];
#pragma unroll
  for (int i = 0; i < 4; i++) res[i] = (cx * hm[i] + cy * bv[i]) / den;
  float rn = fmaxf(sqrtf(wsum(res[0]*res[0]+res[1]*res[1]+res[2]*res[2]+res[3]*res[3])), MINF);
  float pf = (rn > MAXNRM) ? (MAXNRM / rn) : 1.f;
  float nn = fminf(rn, MAXNRM);
  float lf = atanhf(fminf(nn, ATCLIP)) / nn * pf;  // logmap0 of proj'ed point
  float u[4], av[4], rv[4];
#pragma unroll
  for (int i = 0; i < 4; i++) {
    u[i]  = lf * res[i];
    av[i] = al[d0 + i];
    rv[i] = ar[d0 + i];
  }
  float elv = wsum(u[0]*av[0] + u[1]*av[1] + u[2]*av[2] + u[3]*av[3]);
  float erv = wsum(u[0]*rv[0] + u[1]*rv[1] + u[2]*rv[2] + u[3]*rv[3]);
  if (lane == 0) { el[row] = elv; er[row] = erv; }
#pragma unroll
  for (int i = 0; i < 4; i++)
    xtT[(size_t)(d0 + i) * NN + row] = (_Float16)u[i];
}

// ----- fused masked softmax + aggregation, column-split, online softmax -----
// one wave = 16 rows x one column split; 8 WMMAs per 32-column step
__global__ void __launch_bounds__(128) hgat_attn(const float* __restrict__ adj,
                                                 const float* __restrict__ el,
                                                 const float* __restrict__ er,
                                                 const _Float16* __restrict__ xtT,
                                                 float* __restrict__ pacc,
                                                 float* __restrict__ pm,
                                                 float* __restrict__ pl) {
  int lane = threadIdx.x & 31, wid = threadIdx.x >> 5;
  int rb    = (blockIdx.x * 4 + wid) * 16;
  int split = blockIdx.y;
  int m  = lane & 15;
  int hi = (lane & 16) ? 1 : 0;
  int koff = hi ? 8 : 0;
  float elm = el[rb + m];
  float mrun = NEGINF, lsum = 0.f;
  v8f acc[8];
  v8f zero = {0.f,0.f,0.f,0.f,0.f,0.f,0.f,0.f};
#pragma unroll
  for (int t = 0; t < 8; t++) acc[t] = zero;

  const float* achunk = adj + (size_t)(rb + m) * NN + (size_t)split * CHUNK;
  const float* echunk = er + (size_t)split * CHUNK;

  for (int cbi = 0; cbi < CHUNK; cbi += 32) {
    const float* ap = achunk + cbi + koff;
    const float* ep = echunk + cbi + koff;
    if (cbi + 32 < CHUNK) __builtin_prefetch((const void*)(ap + 32), 0, 0);

    float lg[16];
#pragma unroll
    for (int i = 0; i < 8; i++) {
      float a0 = ap[i],      a1 = ap[16 + i];
      float v0 = elm + ep[i],      v1 = elm + ep[16 + i];
      float e0 = v0 > 0.f ? v0 : 0.2f * v0;
      float e1 = v1 > 0.f ? v1 : 0.2f * v1;
      lg[i]     = a0 > 0.f ? e0 : NEGINF;
      lg[8 + i] = a1 > 0.f ? e1 : NEGINF;
    }
    float tm = lg[0];
#pragma unroll
    for (int i = 1; i < 16; i++) tm = fmaxf(tm, lg[i]);
    tm = fmaxf(tm, __shfl_xor(tm, 16, 32));
    float mnew  = fmaxf(mrun, tm);
    float alpha = __expf(mrun - mnew);
    mrun = mnew;

    union { v16h v; _Float16 e[16]; } pa;
    float psum = 0.f;
#pragma unroll
    for (int i = 0; i < 16; i++) {
      float p = __expf(lg[i] - mnew);
      psum += p;
      pa.e[i] = (_Float16)p;
    }
    psum += __shfl_xor(psum, 16, 32);
    lsum = lsum * alpha + psum;

    float am[8];
#pragma unroll
    for (int j = 0; j < 8; j++) am[j] = __shfl(alpha, hi ? (j + 8) : j, 32);

    int kb = split * CHUNK + cbi + (hi ? 8 : 0);
#pragma unroll
    for (int t = 0; t < 8; t++) {
      v8f c = acc[t];
#pragma unroll
      for (int j = 0; j < 8; j++) c[j] *= am[j];
      v16h bm = ldA16(xtT + (size_t)(t * 16 + m) * NN + kb);
      acc[t] = __builtin_amdgcn_wmma_f32_16x16x32_f16(false, pa.v, false, bm, (short)0, c, false, false);
    }
  }

  if (lane < 16) {
    pm[(size_t)split * NN + rb + m] = mrun;
    pl[(size_t)split * NN + rb + m] = lsum;
  }
  int mh = hi ? 8 : 0;
#pragma unroll
  for (int t = 0; t < 8; t++)
#pragma unroll
    for (int j = 0; j < 8; j++)
      pacc[((size_t)split * NN + rb + j + mh) * DD + t * 16 + m] = acc[t][j];
}

// ---------- merge column-split softmax partials: agg = att @ xt -------------
__global__ void hgat_combine(const float* __restrict__ pacc, const float* __restrict__ pm,
                             const float* __restrict__ pl, float* __restrict__ agg) {
  int lane = threadIdx.x & 31, wid = threadIdx.x >> 5;
  int row = blockIdx.x * 4 + wid;
  float mv[SPLIT], wv[SPLIT];
  float M = -3e38f;
#pragma unroll
  for (int s = 0; s < SPLIT; s++) { mv[s] = pm[(size_t)s * NN + row]; M = fmaxf(M, mv[s]); }
  float L = 0.f;
#pragma unroll
  for (int s = 0; s < SPLIT; s++) {
    wv[s] = __expf(mv[s] - M);
    L += pl[(size_t)s * NN + row] * wv[s];
  }
  L = fmaxf(L, 1e-30f);
  int d0 = lane * 4;
  float o[4] = {0.f, 0.f, 0.f, 0.f};
#pragma unroll
  for (int s = 0; s < SPLIT; s++) {
    const float* pr = pacc + ((size_t)s * NN + row) * DD + d0;
#pragma unroll
    for (int i = 0; i < 4; i++) o[i] += pr[i] * wv[s];
  }
  float* ag = agg + (size_t)row * DD + d0;
#pragma unroll
  for (int i = 0; i < 4; i++) ag[i] = o[i] / L;
}

// ---- epilogue: proj(expmap0(agg)) then proj(expmap0(relu(logmap0(.)))) -----
__global__ void hgat_epilogue(const float* __restrict__ agg, float* __restrict__ out,
                              _Float16* __restrict__ out16) {
  int lane = threadIdx.x & 31, wid = threadIdx.x >> 5;
  int row = blockIdx.x * 4 + wid;
  const float* gr = agg + (size_t)row * DD + lane * 4;
  float g[4] = {gr[0], gr[1], gr[2], gr[3]};
  float n = fmaxf(sqrtf(wsum(g[0]*g[0]+g[1]*g[1]+g[2]*g[2]+g[3]*g[3])), MINF);
  float t = fminf(tanhf(n), MAXNRM);          // norm of proj(expmap0(agg))
  float f = t / n;
  float tt = fmaxf(t, MINF);
  float lf = atanhf(fminf(tt, ATCLIP)) / tt * f;   // u = lf * g (tangent vec)
  float u[4];
#pragma unroll
  for (int i = 0; i < 4; i++) u[i] = fmaxf(lf * g[i], 0.f);    // relu
  float n2 = fmaxf(sqrtf(wsum(u[0]*u[0]+u[1]*u[1]+u[2]*u[2]+u[3]*u[3])), MINF);
  float f2 = fminf(tanhf(n2), MAXNRM) / n2;
  float* orow = out + (size_t)row * DD + lane * 4;
  if (out16) {
    h4 o;
#pragma unroll
    for (int i = 0; i < 4; i++) { float r = f2 * u[i]; orow[i] = r; o[i] = (_Float16)r; }
    *(h4*)(out16 + (size_t)row * DD + lane * 4) = o;
  } else {
#pragma unroll
    for (int i = 0; i < 4; i++) orow[i] = f2 * u[i];
  }
}

// ---------------------------------------------------------------------------
extern "C" void kernel_launch(void* const* d_in, const int* in_sizes, int n_in,
                              void* d_out, int out_size, void* d_ws, size_t ws_size,
                              hipStream_t stream) {
  (void)in_sizes; (void)n_in; (void)out_size; (void)ws_size;
  const float* x   = (const float*)d_in[0];
  const float* adj = (const float*)d_in[1];
  const float* W1  = (const float*)d_in[2];
  const float* b1  = (const float*)d_in[3];
  const float* al1 = (const float*)d_in[4];
  const float* ar1 = (const float*)d_in[5];
  const float* W2  = (const float*)d_in[6];
  const float* b2  = (const float*)d_in[7];
  const float* al2 = (const float*)d_in[8];
  const float* ar2 = (const float*)d_in[9];

  char* w = (char*)d_ws;
  size_t off = 0;
  auto alloc = [&](size_t bytes) -> void* {
    void* p = w + off;
    off += (bytes + 255) & ~(size_t)255;
    return p;
  };
  float*    hA   = (float*)alloc((size_t)NN * DD * 4);
  _Float16* hA16 = (_Float16*)alloc((size_t)NN * DD * 2);
  float*    mx   = (float*)alloc((size_t)NN * DD * 4);   // also reused as agg
  _Float16* xtT  = (_Float16*)alloc((size_t)NN * DD * 2);
  float*    el   = (float*)alloc((size_t)NN * 4);
  float*    er   = (float*)alloc((size_t)NN * 4);
  float*    bh   = (float*)alloc((size_t)(DD + 1) * 4);
  _Float16* wf16 = (_Float16*)alloc((size_t)DD * DD * 2);
  float*    hB   = (float*)alloc((size_t)NN * DD * 4);
  _Float16* hB16 = (_Float16*)alloc((size_t)NN * DD * 2);
  float*    pacc = (float*)alloc((size_t)SPLIT * NN * DD * 4);
  float*    pm   = (float*)alloc((size_t)SPLIT * NN * 4);
  float*    pl   = (float*)alloc((size_t)SPLIT * NN * 4);
  float*    agg  = mx;   // mx is consumed by rowops before combine writes agg

  dim3 attnGrid(NN / 64, SPLIT);

  hgat_encode<<<NN / 4, 128, 0, stream>>>(x, hA, hA16);

  // ---- layer 1 ----
  hgat_prep<<<1, DD, 0, stream>>>(W1, b1, wf16, bh);
  hgat_gemm<<<NN / 16, 256, 0, stream>>>(hA16, wf16, mx);
  hgat_rowops<<<NN / 4, 128, 0, stream>>>(hA, mx, bh, al1, ar1, xtT, el, er);
  hgat_attn<<<attnGrid, 128, 0, stream>>>(adj, el, er, xtT, pacc, pm, pl);
  hgat_combine<<<NN / 4, 128, 0, stream>>>(pacc, pm, pl, agg);
  hgat_epilogue<<<NN / 4, 128, 0, stream>>>(agg, hB, hB16);

  // ---- layer 2 ----
  hgat_prep<<<1, DD, 0, stream>>>(W2, b2, wf16, bh);
  hgat_gemm<<<NN / 16, 256, 0, stream>>>(hB16, wf16, mx);
  hgat_rowops<<<NN / 4, 128, 0, stream>>>(hB, mx, bh, al2, ar2, xtT, el, er);
  hgat_attn<<<attnGrid, 128, 0, stream>>>(adj, el, er, xtT, pacc, pm, pl);
  hgat_combine<<<NN / 4, 128, 0, stream>>>(pacc, pm, pl, agg);
  hgat_epilogue<<<NN / 4, 128, 0, stream>>>(agg, (float*)d_out, (_Float16*)nullptr);
}
